// ConvSNN5_50706383896822
// MI455X (gfx1250) — matrix operand order repaired and checked by reading, not compiled, not run on hardware
//
#include <hip/hip_runtime.h>

// ---- vector types for WMMA fragments -------------------------------------
typedef _Float16 h8   __attribute__((ext_vector_type(8)));
typedef _Float16 v16h __attribute__((ext_vector_type(16)));
typedef float    v8f  __attribute__((ext_vector_type(8)));

#define DEV static __device__ __forceinline__

// LIF constants: v_dec = 0.9 v + 0.1 i ; i_dec = 0.8 i
#define VDEC_A 0.9f
#define VDEC_B 0.1f
#define IDEC   0.8f

DEV v8f wmma16(v16h a, v16h b, v8f c) {
    return __builtin_amdgcn_wmma_f32_16x16x32_f16(false, a, false, b,
                                                  (short)0, c, false, false);
}

DEV v8f zero8() {
    v8f c;
#pragma unroll
    for (int r = 0; r < 8; ++r) c[r] = 0.0f;
    return c;
}

// A fragment (16x32 f16): lane m=lane&15 = row; g=lane>>4 selects
// K range {g*16..g*16+15} (contiguous). p points at row m's K=0.
DEV v16h load_a16(const _Float16* p, int g) {
    h8 lo = *(const h8*)(p + g * 16);
    h8 hi = *(const h8*)(p + g * 16 + 8);
    v16h a;
#pragma unroll
    for (int j = 0; j < 8; ++j) { a[j] = lo[j]; a[j + 8] = hi[j]; }
    return a;
}

// A gather with zero padding at image borders (channels-last [h][w][C]).
DEV v16h a_gather(const _Float16* img, int sh, int sw, int H, int W, int C,
                  int cbase, int g) {
    v16h a;
    if (sh >= 0 && sh < H && sw >= 0 && sw < W) {
        const _Float16* p = img + ((size_t)(sh * W + sw)) * C + cbase;
        h8 lo = *(const h8*)(p + g * 16);
        h8 hi = *(const h8*)(p + g * 16 + 8);
#pragma unroll
        for (int j = 0; j < 8; ++j) { a[j] = lo[j]; a[j + 8] = hi[j]; }
    } else {
#pragma unroll
        for (int j = 0; j < 16; ++j) a[j] = (_Float16)0.0f;
    }
    return a;
}

// B fragment pre-packed as 32 lanes x 16 halfs contiguous.
DEV v16h load_bfrag(const _Float16* p) {
    h8 lo = *(const h8*)(p);
    h8 hi = *(const h8*)(p + 8);
    v16h b;
#pragma unroll
    for (int j = 0; j < 8; ++j) { b[j] = lo[j]; b[j + 8] = hi[j]; }
    return b;
}

// ---- zero init of LIF/LI states ------------------------------------------
__global__ void snn_zero(float* __restrict__ p, size_t n) {
    size_t i = (size_t)blockIdx.x * blockDim.x + threadIdx.x;
    size_t stride = (size_t)gridDim.x * blockDim.x;
    for (; i < n; i += stride) p[i] = 0.0f;
}

// ---- weight packing into B-fragment order --------------------------------
// B fragment mapping (contiguous K, same order as load_a16): lane n=lane&15, g=lane>>4, K = g*16+j.
__global__ void snn_prep_w2(const float* __restrict__ w2, _Float16* __restrict__ w2f) {
    int idx = blockIdx.x * blockDim.x + threadIdx.x;        // 9*4*512
    if (idx >= 9 * 4 * 512) return;
    int j = idx & 15, lane = (idx >> 4) & 31, blk = idx >> 9;
    int ntile = blk & 3, tap = blk >> 2;
    int g = lane >> 4, nl = lane & 15;
    int k = g * 16 + j;                 // ci
    int co = ntile * 16 + nl;
    w2f[idx] = (_Float16)w2[(co * 32 + k) * 9 + tap];
}

__global__ void snn_prep_w3(const float* __restrict__ w3, _Float16* __restrict__ w3f) {
    int idx = blockIdx.x * blockDim.x + threadIdx.x;        // 9*2*8*512
    if (idx >= 9 * 2 * 8 * 512) return;
    int j = idx & 15, lane = (idx >> 4) & 31, blk = idx >> 9;
    int ntile = blk & 7, rr = blk >> 3;
    int ks = rr & 1, tap = rr >> 1;
    int g = lane >> 4, nl = lane & 15;
    int k = ks * 32 + g * 16 + j;       // ci
    int co = ntile * 16 + nl;
    w3f[idx] = (_Float16)w3[(co * 64 + k) * 9 + tap];
}

__global__ void snn_prep_wfc1(const float* __restrict__ w, _Float16* __restrict__ wf) {
    int idx = blockIdx.x * blockDim.x + threadIdx.x;        // 64*64*512
    if (idx >= 64 * 64 * 512) return;
    int j = idx & 15, lane = (idx >> 4) & 31, blk = idx >> 9;
    int kt = blk & 63, nt = blk >> 6;
    int g = lane >> 4, nl = lane & 15;
    int k = kt * 32 + g * 16 + j;
    int n = nt * 16 + nl;
    wf[idx] = (_Float16)w[(size_t)n * 2048 + k];
}

__global__ void snn_prep_wout(const float* __restrict__ w, _Float16* __restrict__ wf) {
    int idx = blockIdx.x * blockDim.x + threadIdx.x;        // 32*512
    if (idx >= 32 * 512) return;
    int j = idx & 15, lane = (idx >> 4) & 31, kt = idx >> 9;
    int g = lane >> 4, nl = lane & 15;
    int k = kt * 32 + g * 16 + j;
    wf[idx] = (nl < 10) ? (_Float16)w[nl * 1024 + k] : (_Float16)0.0f;
}

// ---- layer 1: conv3x3 (3->32) fp32 VALU + LIF + avgpool -------------------
// blockIdx = b*32 + co (uniform per block -> weight reads become scalar loads)
// thread = one 2x2 pool cell; loads one 4x4 patch per input channel.
__global__ void __launch_bounds__(256)
snn_conv1(const float* __restrict__ x, const float* __restrict__ w1,
          float* __restrict__ v1, float* __restrict__ i1,
          _Float16* __restrict__ spk1, int t) {
    int tid = threadIdx.x;                  // 256 = 16x16 pool cells
    int pw = tid & 15, ph = tid >> 4;
    int co = blockIdx.x & 31;               // uniform
    int b  = blockIdx.x >> 5;               // uniform
    float w[27];
#pragma unroll
    for (int q = 0; q < 27; ++q) w[q] = w1[co * 27 + q];   // scalar loads
    const float* xb = x + (((size_t)t * 128 + b) * 3) * 1024;
    int h0 = 2 * ph, w0 = 2 * pw;

    float acc[2][2] = {{0.0f, 0.0f}, {0.0f, 0.0f}};
#pragma unroll
    for (int ci = 0; ci < 3; ++ci) {
        float patch[4][4];
#pragma unroll
        for (int py = 0; py < 4; ++py) {
            int sh = h0 - 1 + py;
#pragma unroll
            for (int px = 0; px < 4; ++px) {
                int sw = w0 - 1 + px;
                patch[py][px] = (sh >= 0 && sh < 32 && sw >= 0 && sw < 32)
                                    ? xb[ci * 1024 + sh * 32 + sw] : 0.0f;
            }
        }
#pragma unroll
        for (int dh = 0; dh < 2; ++dh)
#pragma unroll
            for (int dw = 0; dw < 2; ++dw)
#pragma unroll
                for (int ky = 0; ky < 3; ++ky)
#pragma unroll
                    for (int kx = 0; kx < 3; ++kx)
                        acc[dh][dw] += patch[dh + ky][dw + kx] * w[(ci * 3 + ky) * 3 + kx];
    }

    float zsum = 0.0f;
#pragma unroll
    for (int dh = 0; dh < 2; ++dh) {
#pragma unroll
        for (int dw = 0; dw < 2; ++dw) {
            int h = h0 + dh, ww = w0 + dw;
            size_t si = ((size_t)((b * 32 + co) * 32 + h)) * 32 + ww;
            float v = v1[si], cur = i1[si];
            float vd = VDEC_A * v + VDEC_B * cur;
            float z = (vd > 0.25f) ? 1.0f : 0.0f;
            v1[si] = (1.0f - z) * vd;
            i1[si] = IDEC * cur + acc[dh][dw];
            zsum += z;
        }
    }
    spk1[((size_t)((b * 16 + ph) * 16 + pw)) * 32 + co] = (_Float16)(0.25f * zsum);
}

// ---- layer 2: conv3x3 (32->64) WMMA + LIF + avgpool -----------------------
// block = 4 waves (ntile 0..3); blockIdx = b*8 + rowpair
__global__ void __launch_bounds__(128)
snn_conv2(const _Float16* __restrict__ spk1, const _Float16* __restrict__ w2f,
          float* __restrict__ v2, float* __restrict__ i2,
          _Float16* __restrict__ spk2) {
    int ntile = threadIdx.x >> 5;
    int lane  = threadIdx.x & 31;
    int rp = blockIdx.x & 7, b = blockIdx.x >> 3;
    int g = lane >> 4, nl = lane & 15;
    int h0 = rp * 2;
    const _Float16* img = spk1 + (size_t)b * 16 * 16 * 32;

    v8f c0 = zero8(), c1 = zero8();
#pragma unroll
    for (int tap = 0; tap < 9; ++tap) {
        int ty = tap / 3 - 1, tx = tap % 3 - 1;
        v16h a0 = a_gather(img, h0 + ty,     nl + tx, 16, 16, 32, 0, g);
        v16h a1 = a_gather(img, h0 + 1 + ty, nl + tx, 16, 16, 32, 0, g);
        v16h bf = load_bfrag(w2f + (size_t)((tap * 4 + ntile) * 512) + lane * 16);
        c0 = wmma16(a0, bf, c0);
        c1 = wmma16(a1, bf, c1);
    }
    int co = ntile * 16 + nl;
    float z0[8], z1[8];
#pragma unroll
    for (int r = 0; r < 8; ++r) {
        int w = r + 8 * g;
        {
            size_t si = ((size_t)((b * 16 + h0) * 16 + w)) * 64 + co;
            float v = v2[si], cur = i2[si];
            float vd = VDEC_A * v + VDEC_B * cur;
            float z = (vd > 0.2f) ? 1.0f : 0.0f;
            v2[si] = (1.0f - z) * vd;
            i2[si] = IDEC * cur + c0[r];
            z0[r] = z;
        }
        {
            size_t si = ((size_t)((b * 16 + h0 + 1) * 16 + w)) * 64 + co;
            float v = v2[si], cur = i2[si];
            float vd = VDEC_A * v + VDEC_B * cur;
            float z = (vd > 0.2f) ? 1.0f : 0.0f;
            v2[si] = (1.0f - z) * vd;
            i2[si] = IDEC * cur + c1[r];
            z1[r] = z;
        }
    }
#pragma unroll
    for (int q = 0; q < 4; ++q) {
        float s = z0[2 * q] + z0[2 * q + 1] + z1[2 * q] + z1[2 * q + 1];
        int pw = 4 * g + q;
        spk2[((size_t)((b * 8 + rp) * 8 + pw)) * 64 + co] = (_Float16)(0.25f * s);
    }
}

// ---- layer 3: conv3x3 (64->128) WMMA + LIF + avgpool ----------------------
// block = 8 waves (ntile 0..7); blockIdx = b*4 + rowpair; M-tile = 2 rows x 8 cols
__global__ void __launch_bounds__(256)
snn_conv3(const _Float16* __restrict__ spk2, const _Float16* __restrict__ w3f,
          float* __restrict__ v3, float* __restrict__ i3,
          _Float16* __restrict__ fcin) {
    int ntile = threadIdx.x >> 5;
    int lane  = threadIdx.x & 31;
    int rp = blockIdx.x & 3, b = blockIdx.x >> 2;
    int g = lane >> 4, nl = lane & 15;
    int h0 = rp * 2;
    int mrow = nl >> 3, mw = nl & 7;       // A row: pixel (h0+mrow, mw)
    const _Float16* img = spk2 + (size_t)b * 8 * 8 * 64;

    v8f c = zero8();
#pragma unroll
    for (int tap = 0; tap < 9; ++tap) {
        int ty = tap / 3 - 1, tx = tap % 3 - 1;
        int sh = h0 + mrow + ty, sw = mw + tx;
#pragma unroll
        for (int ks = 0; ks < 2; ++ks) {
            v16h a  = a_gather(img, sh, sw, 8, 8, 64, ks * 32, g);
            v16h bf = load_bfrag(w3f + (size_t)(((tap * 2 + ks) * 8 + ntile) * 512) + lane * 16);
            c = wmma16(a, bf, c);
        }
    }
    int co = ntile * 16 + nl;
    int h = h0 + g;                        // lane group selects the row
    float z[8];
#pragma unroll
    for (int r = 0; r < 8; ++r) {
        size_t si = ((size_t)((b * 8 + h) * 8 + r)) * 128 + co;
        float v = v3[si], cur = i3[si];
        float vd = VDEC_A * v + VDEC_B * cur;
        float zz = (vd > 0.1f) ? 1.0f : 0.0f;
        v3[si] = (1.0f - zz) * vd;
        i3[si] = IDEC * cur + c[r];
        z[r] = zz;
    }
#pragma unroll
    for (int q = 0; q < 4; ++q) {
        float s = z[2 * q] + z[2 * q + 1];
        s += __shfl_xor(s, 16);            // combine rows h0 / h0+1
        if (g == 0) {
            // NCHW flatten order: c*16 + h*4 + w  (pooled 4x4)
            fcin[(size_t)b * 2048 + co * 16 + rp * 4 + q] = (_Float16)(0.25f * s);
        }
    }
}

// ---- fc1 (2048->1024) WMMA + LIF ------------------------------------------
// block = 8 waves; blockIdx = mt*8 + ntgrp ; nt = ntgrp*8 + wave
__global__ void __launch_bounds__(256)
snn_fc1(const _Float16* __restrict__ fcin, const _Float16* __restrict__ wfc1f,
        float* __restrict__ v4, float* __restrict__ i4,
        _Float16* __restrict__ z4) {
    int wave = threadIdx.x >> 5;
    int lane = threadIdx.x & 31;
    int g = lane >> 4, nl = lane & 15;
    int mt = blockIdx.x >> 3;
    int nt = (blockIdx.x & 7) * 8 + wave;

    const _Float16* arow  = fcin + (size_t)(mt * 16 + nl) * 2048;
    const _Float16* bbase = wfc1f + (size_t)(nt * 64) * 512 + lane * 16;
    v8f c = zero8();
#pragma unroll 4
    for (int kt = 0; kt < 64; ++kt) {
        v16h a  = load_a16(arow + kt * 32, g);
        v16h bf = load_bfrag(bbase + (size_t)kt * 512);
        c = wmma16(a, bf, c);
    }
    int n = nt * 16 + nl;
#pragma unroll
    for (int r = 0; r < 8; ++r) {
        int bb = mt * 16 + r + 8 * g;
        size_t si = (size_t)bb * 1024 + n;
        float v = v4[si], cur = i4[si];
        float vd = VDEC_A * v + VDEC_B * cur;
        float zz = (vd > 0.1f) ? 1.0f : 0.0f;
        v4[si] = (1.0f - zz) * vd;
        i4[si] = IDEC * cur + c[r];
        z4[si] = (_Float16)zz;
    }
}

// ---- fc_out (1024->10) WMMA + LI + running max ----------------------------
// 1 block, 8 waves = 8 M-tiles
__global__ void __launch_bounds__(256)
snn_fc2(const _Float16* __restrict__ z4, const _Float16* __restrict__ woutf,
        float* __restrict__ vo, float* __restrict__ io,
        float* __restrict__ out, int t) {
    int mt   = threadIdx.x >> 5;
    int lane = threadIdx.x & 31;
    int g = lane >> 4, nl = lane & 15;

    const _Float16* arow = z4 + (size_t)(mt * 16 + nl) * 1024;
    v8f c = zero8();
#pragma unroll 4
    for (int kt = 0; kt < 32; ++kt) {
        v16h a  = load_a16(arow + kt * 32, g);
        v16h bf = load_bfrag(woutf + (size_t)kt * 512 + lane * 16);
        c = wmma16(a, bf, c);
    }
    if (nl < 10) {
#pragma unroll
        for (int r = 0; r < 8; ++r) {
            int bb = mt * 16 + r + 8 * g;
            size_t si = (size_t)bb * 10 + nl;
            float v = vo[si], cur = io[si];
            float vn = VDEC_A * v + VDEC_B * cur;   // li_step voltage
            io[si] = IDEC * cur + c[r];
            vo[si] = vn;
            out[si] = (t == 0) ? vn : fmaxf(out[si], vn);
        }
    }
}

// ---- host-side launch -----------------------------------------------------
extern "C" void kernel_launch(void* const* d_in, const int* in_sizes, int n_in,
                              void* d_out, int out_size, void* d_ws, size_t ws_size,
                              hipStream_t stream) {
    (void)in_sizes; (void)n_in; (void)out_size; (void)ws_size;
    const float* x    = (const float*)d_in[0];
    const float* w1   = (const float*)d_in[1];
    const float* w2   = (const float*)d_in[2];
    const float* w3   = (const float*)d_in[3];
    const float* wfc1 = (const float*)d_in[4];
    const float* wout = (const float*)d_in[5];
    float* out = (float*)d_out;

    char* ws = (char*)d_ws;
    size_t off = 0;
    auto alloc = [&](size_t bytes) -> void* {
        void* p = ws + off;
        off = (off + bytes + 255) & ~(size_t)255;
        return p;
    };

    const size_t N1 = 128u * 32 * 32 * 32;   // 4,194,304
    const size_t N2 = 128u * 64 * 16 * 16;   // 2,097,152
    const size_t N3 = 128u * 128 * 8 * 8;    // 1,048,576
    const size_t N4 = 128u * 1024;           // 131,072
    const size_t NO = 128u * 10;             // 1,280
    const size_t nState = 2 * (N1 + N2 + N3 + N4 + NO);

    float* stateBase = (float*)alloc(nState * sizeof(float));
    float* v1 = stateBase;       float* i1 = v1 + N1;
    float* v2 = i1 + N1;         float* i2 = v2 + N2;
    float* v3 = i2 + N2;         float* i3 = v3 + N3;
    float* v4 = i3 + N3;         float* i4 = v4 + N4;
    float* vo = i4 + N4;         float* io = vo + NO;

    _Float16* spk1  = (_Float16*)alloc((size_t)128 * 16 * 16 * 32 * 2);
    _Float16* spk2  = (_Float16*)alloc((size_t)128 * 8 * 8 * 64 * 2);
    _Float16* fcin  = (_Float16*)alloc((size_t)128 * 2048 * 2);
    _Float16* z4    = (_Float16*)alloc((size_t)128 * 1024 * 2);
    _Float16* w2f   = (_Float16*)alloc((size_t)9 * 4 * 512 * 2);
    _Float16* w3f   = (_Float16*)alloc((size_t)9 * 2 * 8 * 512 * 2);
    _Float16* wfc1f = (_Float16*)alloc((size_t)64 * 64 * 512 * 2);
    _Float16* woutf = (_Float16*)alloc((size_t)32 * 512 * 2);

    snn_zero<<<4096, 256, 0, stream>>>(stateBase, nState);
    snn_prep_w2<<<(9 * 4 * 512 + 255) / 256, 256, 0, stream>>>(w2, w2f);
    snn_prep_w3<<<(9 * 2 * 8 * 512 + 255) / 256, 256, 0, stream>>>(w3, w3f);
    snn_prep_wfc1<<<(64 * 64 * 512) / 256, 256, 0, stream>>>(wfc1, wfc1f);
    snn_prep_wout<<<(32 * 512) / 256, 256, 0, stream>>>(wout, woutf);

    for (int t = 0; t < 16; ++t) {
        snn_conv1<<<128 * 32, 256, 0, stream>>>(x, w1, v1, i1, spk1, t);
        snn_conv2<<<128 * 8, 128, 0, stream>>>(spk1, w2f, v2, i2, spk2);
        snn_conv3<<<128 * 4, 256, 0, stream>>>(spk2, w3f, v3, i3, fcin);
        snn_fc1<<<64, 256, 0, stream>>>(fcin, wfc1f, v4, i4, z4);
        snn_fc2<<<1, 256, 0, stream>>>(z4, woutf, vo, io, out, t);
    }
}
